// MultiHeadAttention_22462678958551
// MI455X (gfx1250) — compile-verified, matrix operand order checked by reading, not run on hardware
//
#include <hip/hip_runtime.h>

// ---------------------------------------------------------------------------
// MHA forward for MI455X (gfx1250, wave32). All matmuls via
// v_wmma_f32_16x16x32_bf16; register double-buffered operand streaming;
// intermediates stay L2-resident (192 MB L2 >> 48 MB working set).
// ---------------------------------------------------------------------------

typedef __attribute__((ext_vector_type(16))) __bf16 v16bf;
typedef __attribute__((ext_vector_type(8)))  __bf16 bf16x8;
typedef __attribute__((ext_vector_type(8)))  float  v8f;

using u16 = unsigned short;

#define DEV __device__ __forceinline__

#define BB     2
#define LL     2048
#define MTOK   4096              // B*L tokens

// round-to-nearest-even f32 -> bf16
DEV u16 f2bf(float f) {
  unsigned int u = __builtin_bit_cast(unsigned int, f);
  u = (u + 0x7FFFu + ((u >> 16) & 1u)) >> 16;
  return (u16)u;
}

// 16x32 bf16 WMMA fragment from two contiguous 16B chunks (ISA 16-bit layout:
// lane half selects K-chunk pair {off, off+16}, off = (lane>>4)*8).
DEV v16bf load_frag(const u16* p0, const u16* p1) {
  bf16x8 a = *reinterpret_cast<const bf16x8*>(p0);
  bf16x8 b = *reinterpret_cast<const bf16x8*>(p1);
  v16bf r;
#pragma unroll
  for (int i = 0; i < 8; ++i) { r[i] = a[i]; r[i + 8] = b[i]; }
  return r;
}

DEV v8f wmma_bf16(v16bf a, v16bf b, v8f c) {
  return __builtin_amdgcn_wmma_f32_16x16x32_bf16(false, a, false, b,
                                                 (short)0, c, false, false);
}

// xor-butterfly within 16-lane halves via ds_swizzle_b32
template <int XM>
DEV float swz(float v) {
  int i = __builtin_bit_cast(int, v);
  i = __builtin_amdgcn_ds_swizzle(i, (XM << 10) | 0x1F);
  return __builtin_bit_cast(float, i);
}
DEV float rowmax16(float v) {
  v = fmaxf(v, swz<1>(v)); v = fmaxf(v, swz<2>(v));
  v = fmaxf(v, swz<4>(v)); v = fmaxf(v, swz<8>(v));
  return v;
}
DEV float rowsum16(float v) {
  v += swz<1>(v); v += swz<2>(v); v += swz<4>(v); v += swz<8>(v);
  return v;
}

// ---------------------------------------------------------------------------
// f32 -> bf16, 4 elements/thread, packed 2x16b stores
__global__ __launch_bounds__(256) void cvt_bf16_kernel(
    const float* __restrict__ in, u16* __restrict__ out) {
  int i = (blockIdx.x * 256 + threadIdx.x) * 4;
  float4 v = *reinterpret_cast<const float4*>(in + i);
  uint2 p;
  p.x = (unsigned)f2bf(v.x) | ((unsigned)f2bf(v.y) << 16);
  p.y = (unsigned)f2bf(v.z) | ((unsigned)f2bf(v.w) << 16);
  *reinterpret_cast<uint2*>(out + i) = p;
}

// w[K=1024][N=1024] f32 -> wt[N][K] bf16, LDS-tiled 32x32 for coalescing
__global__ __launch_bounds__(256) void transpose_bf16_kernel(
    const float* __restrict__ w, u16* __restrict__ wt) {
  __shared__ u16 tile[32][33];
  const int tx = threadIdx.x & 31, ty = threadIdx.x >> 5;  // 32 x 8
  const int k0 = blockIdx.y * 32, n0 = blockIdx.x * 32;
#pragma unroll
  for (int r = 0; r < 32; r += 8)
    tile[ty + r][tx] = f2bf(w[(size_t)(k0 + ty + r) * 1024 + n0 + tx]);
  __syncthreads();
#pragma unroll
  for (int r = 0; r < 32; r += 8)
    wt[(size_t)(n0 + ty + r) * 1024 + k0 + tx] = tile[tx][ty + r];
}

// ---------------------------------------------------------------------------
// WMMA GEMM: C[4096, 1024] = A[4096,1024] @ Bt[1024,1024]^T + bias.
// Wave tile 32x64 (2 A-frags x 4 B-frags -> 8 WMMAs per k-step), register
// double-buffered so WMMAs never wait on same-iteration loads.
// mode 0: bf16 Q/K [B,H,L,Dk] | mode 1: bf16 V^T [B,H,Dk,L] | mode 2: f32 out
__global__ __launch_bounds__(256) void gemm_wmma_kernel(
    const u16* __restrict__ A, const u16* __restrict__ Bt,
    const float* __restrict__ bias, void* __restrict__ outp, int mode) {
  const int lane = threadIdx.x & 31;
  const int wave = threadIdx.x >> 5;
  const int tileM = blockIdx.y * 8 + wave;   // 0..127 (32 rows each)
  const int tileN = blockIdx.x;              // 0..15  (64 cols each)
  const int half = lane >> 4, lo = lane & 15, koff = half * 8;

  const v8f zero = {0.f, 0.f, 0.f, 0.f, 0.f, 0.f, 0.f, 0.f};
  v8f acc[2][4] = {{zero, zero, zero, zero}, {zero, zero, zero, zero}};

  const u16* arow0 = A + (size_t)(tileM * 32 + lo) * 1024 + koff;
  const u16* arow1 = arow0 + 16 * 1024;
  const u16* brow[4];
#pragma unroll
  for (int t = 0; t < 4; ++t)
    brow[t] = Bt + (size_t)(tileN * 64 + t * 16 + lo) * 1024 + koff;

  // prologue: fragments for kb = 0
  v16bf a0 = load_frag(arow0, arow0 + 16);
  v16bf a1 = load_frag(arow1, arow1 + 16);
  v16bf b0 = load_frag(brow[0], brow[0] + 16);
  v16bf b1 = load_frag(brow[1], brow[1] + 16);
  v16bf b2 = load_frag(brow[2], brow[2] + 16);
  v16bf b3 = load_frag(brow[3], brow[3] + 16);

  for (int kb = 0; kb < 1024; kb += 32) {
    // issue next k-step's loads first (wrap harmlessly on last iteration)
    const int kn = (kb + 32 < 1024) ? kb + 32 : 0;
    v16bf na0 = load_frag(arow0 + kn, arow0 + kn + 16);
    v16bf na1 = load_frag(arow1 + kn, arow1 + kn + 16);
    v16bf nb0 = load_frag(brow[0] + kn, brow[0] + kn + 16);
    v16bf nb1 = load_frag(brow[1] + kn, brow[1] + kn + 16);
    v16bf nb2 = load_frag(brow[2] + kn, brow[2] + kn + 16);
    v16bf nb3 = load_frag(brow[3] + kn, brow[3] + kn + 16);

    // 8 WMMAs on current fragments (wait only covers prev-iteration loads)
    acc[0][0] = wmma_bf16(a0, b0, acc[0][0]);
    acc[0][1] = wmma_bf16(a0, b1, acc[0][1]);
    acc[0][2] = wmma_bf16(a0, b2, acc[0][2]);
    acc[0][3] = wmma_bf16(a0, b3, acc[0][3]);
    acc[1][0] = wmma_bf16(a1, b0, acc[1][0]);
    acc[1][1] = wmma_bf16(a1, b1, acc[1][1]);
    acc[1][2] = wmma_bf16(a1, b2, acc[1][2]);
    acc[1][3] = wmma_bf16(a1, b3, acc[1][3]);

    a0 = na0; a1 = na1; b0 = nb0; b1 = nb1; b2 = nb2; b3 = nb3;
  }

  // Epilogue. C layout: VGPR i = row (i + 8*half), lane lo = column.
#pragma unroll
  for (int r = 0; r < 2; ++r) {
#pragma unroll
    for (int t = 0; t < 4; ++t) {
      const int col = tileN * 64 + t * 16 + lo;
      const float bv = bias[col];
#pragma unroll
      for (int i = 0; i < 8; ++i) {
        const int m = tileM * 32 + r * 16 + half * 8 + i;   // token
        const float val = acc[r][t][i] + bv;
        if (mode == 2) {
          ((float*)outp)[(size_t)m * 1024 + col] = val;
        } else {
          const int b = m >> 11, l = m & 2047;
          const int h = col >> 6, dk = col & 63;
          const u16 bb = f2bf(val);
          if (mode == 0)       // [B,H,L,Dk]
            ((u16*)outp)[(((size_t)(b * 16 + h) * 2048) + l) * 64 + dk] = bb;
          else                 // [B,H,Dk,L]
            ((u16*)outp)[(((size_t)(b * 16 + h) * 64) + dk) * 2048 + l] = bb;
        }
      }
    }
  }
}

// ---------------------------------------------------------------------------
// Flash attention: one wave per 16-query tile, 32-key steps.
// K fragments double-buffered; V fragments loaded BEFORE the softmax/LDS
// memory-clobber so P@V WMMAs never stall on fresh loads.
__global__ __launch_bounds__(128) void attn_wmma_kernel(
    const u16* __restrict__ Q,   // [BH, L, 64] bf16
    const u16* __restrict__ Km,  // [BH, L, 64] bf16
    const u16* __restrict__ Vt,  // [BH, 64, L] bf16
    u16* __restrict__ ctx) {     // [B, L, INNER] bf16
  __shared__ __align__(16) u16 pst_all[4][16 * 32];
  const int lane = threadIdx.x & 31;
  const int wave = threadIdx.x >> 5;
  const int bh = blockIdx.x;                 // 0..31
  const int qt = blockIdx.y * 4 + wave;      // 0..127
  u16* pst = &pst_all[wave][0];

  const int half = lane >> 4, lo = lane & 15, koff = half * 8;

  const u16* qrow = Q + (size_t)(bh * 2048 + qt * 16 + lo) * 64;
  v16bf qa0 = load_frag(qrow + koff,      qrow + koff + 16);
  v16bf qa1 = load_frag(qrow + 32 + koff, qrow + 32 + koff + 16);

  const v8f zero = {0.f, 0.f, 0.f, 0.f, 0.f, 0.f, 0.f, 0.f};
  v8f o[4] = {zero, zero, zero, zero};
  float mrun[8], lrun[8];
#pragma unroll
  for (int i = 0; i < 8; ++i) { mrun[i] = -3.0e38f; lrun[i] = 0.f; }

  const u16* kbh = Km + (size_t)bh * 2048 * 64;
  const u16* vbh = Vt + (size_t)bh * 64 * 2048;

  // prologue: K fragments for keyBase = 0
  v16bf kc00, kc01, kc10, kc11;
  {
    const u16* kr0 = kbh + (size_t)lo * 64 + koff;
    const u16* kr1 = kbh + (size_t)(16 + lo) * 64 + koff;
    kc00 = load_frag(kr0, kr0 + 16);  kc01 = load_frag(kr0 + 32, kr0 + 48);
    kc10 = load_frag(kr1, kr1 + 16);  kc11 = load_frag(kr1 + 32, kr1 + 48);
  }

  for (int keyBase = 0; keyBase < 2048; keyBase += 32) {
    // scores on current (preloaded) K fragments
    v8f s0 = zero, s1 = zero;
    s0 = wmma_bf16(qa0, kc00, s0); s0 = wmma_bf16(qa1, kc01, s0);
    s1 = wmma_bf16(qa0, kc10, s1); s1 = wmma_bf16(qa1, kc11, s1);

    // V fragments for this key block — issued before softmax so the latency
    // hides under the exp/reduction VALU work
    v16bf vf[4];
#pragma unroll
    for (int t = 0; t < 4; ++t) {
      const u16* vr = vbh + (size_t)(t * 16 + lo) * 2048 + keyBase + koff;
      vf[t] = load_frag(vr, vr + 16);
    }
    // next key block's K fragments (wrap harmlessly on last iteration)
    {
      const int kB = (keyBase + 32 < 2048) ? keyBase + 32 : 0;
      const u16* kr0 = kbh + (size_t)(kB + lo) * 64 + koff;
      const u16* kr1 = kbh + (size_t)(kB + 16 + lo) * 64 + koff;
      kc00 = load_frag(kr0, kr0 + 16);  kc01 = load_frag(kr0 + 32, kr0 + 48);
      kc10 = load_frag(kr1, kr1 + 16);  kc11 = load_frag(kr1 + 32, kr1 + 48);
    }

    // online softmax (row m lives in VGPR i across a 16-lane half)
#pragma unroll
    for (int i = 0; i < 8; ++i) {
      const float a = s0[i] * 0.125f;            // 1/sqrt(Dk)
      const float b = s1[i] * 0.125f;
      float nm = rowmax16(fmaxf(a, b));
      nm = fmaxf(nm, mrun[i]);
      const float corr = __expf(mrun[i] - nm);
      const float p0 = __expf(a - nm);
      const float p1 = __expf(b - nm);
      lrun[i] = lrun[i] * corr + rowsum16(p0 + p1);
      mrun[i] = nm;
      o[0][i] *= corr; o[1][i] *= corr; o[2][i] *= corr; o[3][i] *= corr;
      const int row = half * 8 + i;
      pst[row * 32 + lo]      = f2bf(p0);
      pst[row * 32 + 16 + lo] = f2bf(p1);
    }
    // same-wave LDS exchange (C-layout -> A-layout for P)
    asm volatile("s_wait_dscnt 0" ::: "memory");
    const u16* pr = pst + lo * 32 + koff;
    v16bf pf = load_frag(pr, pr + 16);

#pragma unroll
    for (int t = 0; t < 4; ++t) o[t] = wmma_bf16(pf, vf[t], o[t]);
  }

  float inv[8];
#pragma unroll
  for (int i = 0; i < 8; ++i) inv[i] = 1.0f / lrun[i];
  const int b = bh >> 4, h = bh & 15;
#pragma unroll
  for (int t = 0; t < 4; ++t) {
#pragma unroll
    for (int i = 0; i < 8; ++i) {
      const int tok = qt * 16 + half * 8 + i;
      ctx[((size_t)(b * 2048 + tok)) * 1024 + h * 64 + t * 16 + lo] =
          f2bf(o[t][i] * inv[i]);
    }
  }
}

// ---------------------------------------------------------------------------
extern "C" void kernel_launch(void* const* d_in, const int* in_sizes, int n_in,
                              void* d_out, int out_size, void* d_ws, size_t ws_size,
                              hipStream_t stream) {
  const float* x  = (const float*)d_in[0];
  const float* wq = (const float*)d_in[1];
  const float* bq = (const float*)d_in[2];
  const float* wk = (const float*)d_in[3];
  const float* bk = (const float*)d_in[4];
  const float* wv = (const float*)d_in[5];
  const float* bv = (const float*)d_in[6];
  const float* wo = (const float*)d_in[7];
  const float* bo = (const float*)d_in[8];

  char* ws = (char*)d_ws;
  size_t off = 0;
  auto take = [&](size_t bytes) -> void* {
    void* p = ws + off;
    off += (bytes + 255) & ~size_t(255);
    return p;
  };
  u16* xb   = (u16*)take((size_t)MTOK * 1024 * 2);
  u16* wqT  = (u16*)take((size_t)1024 * 1024 * 2);
  u16* wkT  = (u16*)take((size_t)1024 * 1024 * 2);
  u16* wvT  = (u16*)take((size_t)1024 * 1024 * 2);
  u16* woT  = (u16*)take((size_t)1024 * 1024 * 2);
  u16* Qb   = (u16*)take((size_t)MTOK * 1024 * 2);
  u16* Kb   = (u16*)take((size_t)MTOK * 1024 * 2);
  u16* Vtb  = (u16*)take((size_t)MTOK * 1024 * 2);
  u16* ctxb = (u16*)take((size_t)MTOK * 1024 * 2);

  // 1) precision conversion / weight transposes
  cvt_bf16_kernel<<<(MTOK * 1024) / (256 * 4), 256, 0, stream>>>(x, xb);
  dim3 tgrid(32, 32);
  transpose_bf16_kernel<<<tgrid, 256, 0, stream>>>(wq, wqT);
  transpose_bf16_kernel<<<tgrid, 256, 0, stream>>>(wk, wkT);
  transpose_bf16_kernel<<<tgrid, 256, 0, stream>>>(wv, wvT);
  transpose_bf16_kernel<<<tgrid, 256, 0, stream>>>(wo, woT);

  // 2) QKV projections (WMMA, 32x64 wave tiles, double-buffered)
  dim3 ggrid(16, 16);  // N/64 x M/(32*8)
  gemm_wmma_kernel<<<ggrid, 256, 0, stream>>>(xb, wqT, bq, Qb, 0);
  gemm_wmma_kernel<<<ggrid, 256, 0, stream>>>(xb, wkT, bk, Kb, 0);
  gemm_wmma_kernel<<<ggrid, 256, 0, stream>>>(xb, wvT, bv, Vtb, 1);

  // 3) flash attention (WMMA + online softmax)
  attn_wmma_kernel<<<dim3(32, 32), 128, 0, stream>>>(Qb, Kb, Vtb, ctxb);

  // 4) output projection -> fp32 d_out
  gemm_wmma_kernel<<<ggrid, 256, 0, stream>>>(ctxb, woT, bo, d_out, 2);
}
